// Speller_86406152060997
// MI455X (gfx1250) — matrix operand order repaired and checked by reading, not compiled, not run on hardware
//
#include <hip/hip_runtime.h>
#include <hip/hip_bf16.h>
#include <stdint.h>
#include <stddef.h>

// ---------------- problem dims ----------------
constexpr int Bz = 128, Sz = 400, ENCz = 512;
constexpr int Ez = 256, Hz = 512, Dz = 128, Pz = 128, Vz = 30;
constexpr int Tz = 600;
constexpr int K1 = Ez + Pz + Hz;   // 896 : [embed | ctx | h1]
constexpr int N1 = 4 * Hz;         // 2048
constexpr int K2 = Hz + Dz;        // 640 : [h1n | h2_prev]
constexpr int N2 = 4 * Dz;         // 512

// ---------------- vector types ----------------
typedef __bf16 v16bf __attribute__((ext_vector_type(16)));
typedef __bf16 v8bf  __attribute__((ext_vector_type(8)));
typedef float  v8f   __attribute__((ext_vector_type(8)));
typedef unsigned int u32x4 __attribute__((ext_vector_type(4)));
typedef int i32x8 __attribute__((ext_vector_type(8)));
typedef int i32x4 __attribute__((ext_vector_type(4)));

// ---------------- workspace layout (bytes, 256B aligned) ----------------
constexpr size_t al256(size_t x) { return (x + 255) & ~(size_t)255; }
constexpr size_t OFF_KEYS  = 0;                                         // [B*S, P] f32
constexpr size_t OFF_VALS  = al256(OFF_KEYS  + (size_t)Bz*Sz*Pz*4);     // [B*S, P] f32
constexpr size_t OFF_WCAT1 = al256(OFF_VALS  + (size_t)Bz*Sz*Pz*4);     // [N1, K1] bf16
constexpr size_t OFF_WCAT2 = al256(OFF_WCAT1 + (size_t)N1*K1*2);        // [N2, K2] bf16
constexpr size_t OFF_B1    = al256(OFF_WCAT2 + (size_t)N2*K2*2);        // [N1] f32
constexpr size_t OFF_B2    = al256(OFF_B1    + (size_t)N1*4);           // [N2] f32
constexpr size_t OFF_XCAT  = al256(OFF_B2    + (size_t)N2*4);           // [B, K1] bf16
constexpr size_t OFF_XCAT2 = al256(OFF_XCAT  + (size_t)Bz*K1*2);        // [B, K2] bf16
constexpr size_t OFF_G1    = al256(OFF_XCAT2 + (size_t)Bz*K2*2);        // [B, N1] f32
constexpr size_t OFF_G2    = al256(OFF_G1    + (size_t)Bz*N1*4);        // [B, N2] f32
constexpr size_t OFF_BAR   = al256(OFF_G2    + (size_t)Bz*N2*4);        // 2 ints

// ---------------- WMMA helpers ----------------
__device__ __forceinline__ v8f wmma_bf16(v16bf a, v16bf b, v8f c) {
  return __builtin_amdgcn_wmma_f32_16x16x32_bf16(false, a, false, b, (short)0, c,
                                                 false, false);
}

// A fragment: rows m0..m0+15, K chunk [k0, k0+32) of a row-major bf16 matrix.
// Lanes 0-15 hold K {0..7, 16..23}; lanes 16-31 hold K {8..15, 24..31}.
__device__ __forceinline__ v16bf load_a_bf16(const __bf16* A, int lda, int m0,
                                             int k0, int lane) {
  int lo = lane & 15, hi = lane >> 4;
  const __bf16* row = A + (size_t)(m0 + lo) * lda + k0;
  v8bf p0 = *(const v8bf*)(row + hi * 8);
  v8bf p1 = *(const v8bf*)(row + 16 + hi * 8);
  v16bf a;
#pragma unroll
  for (int i = 0; i < 8; ++i) { a[i] = p0[i]; a[8 + i] = p1[i]; }
  return a;
}

// Same A fragment but source is f32 (convert in registers).
__device__ __forceinline__ v16bf load_a_f32(const float* A, int lda, int m0,
                                            int k0, int lane) {
  int lo = lane & 15, hi = lane >> 4;
  const float* row = A + (size_t)(m0 + lo) * lda + k0;
  v16bf a;
#pragma unroll
  for (int i = 0; i < 8; ++i) {
    a[i]     = (__bf16)row[hi * 8 + i];
    a[8 + i] = (__bf16)row[16 + hi * 8 + i];
  }
  return a;
}

// B fragment for out = X @ W^T (W row-major [N,K]); lane n%16 owns column,
// lanes 0-15 hold K {0..7, 16..23}; lanes 16-31 hold K {8..15, 24..31}.
__device__ __forceinline__ v16bf load_b_rows(const __bf16* Wrows, int ldw,
                                             int k0, int lane) {
  const __bf16* row = Wrows + (size_t)(lane & 15) * ldw + k0 + ((lane >> 4) * 8);
  v8bf p0 = *(const v8bf*)(row);
  v8bf p1 = *(const v8bf*)(row + 16);
  v16bf b;
#pragma unroll
  for (int i = 0; i < 8; ++i) { b[i] = p0[i]; b[8 + i] = p1[i]; }
  return b;
}

__device__ __forceinline__ v16bf load_b_wf32(const float* W, int ldw, int n0,
                                             int k0, int lane) {
  const float* row = W + (size_t)(n0 + (lane & 15)) * ldw + k0 + ((lane >> 4) * 8);
  v16bf b;
#pragma unroll
  for (int i = 0; i < 16; ++i) b[i] = (__bf16)row[(i & 7) + ((i >> 3) * 16)];
  return b;
}

__device__ __forceinline__ float sigf(float x) { return 1.0f / (1.0f + __expf(-x)); }

// LDS byte offset of a shared-memory object (AS3 pointers are LDS offsets).
__device__ __forceinline__ uint32_t lds_offset(const void* p) {
  return (uint32_t)(uintptr_t)(__attribute__((address_space(3))) const char*)p;
}

// ---- TDM: DMA a [nrows x rowlen] contiguous bf16 slab from global into LDS.
// Builds a 2D D# per CDNA5 ISA ch.8: group0 = {count=1, lds_addr, global_addr,
// type=2}; group1 = {data_size=2B, tensor_dim0=rowlen, tensor_dim1=nrows,
// tile_dim0=rowlen, tile_dim1=nrows, tensor_dim0_stride=rowlen}; groups 2/3
// zero (<=2D tensor). One wave issues, waits TENSORcnt, then all sync.
// This toolchain's builtin is the 6-arg clang-23 form:
//   (u32x4 g0, i32x8 g1, i32x4, i32x4, i32x8, i32 cpol)
__device__ __forceinline__ void stage_rows_lds(const __bf16* gsrc, __bf16* ldst,
                                               int rowlen, int nrows) {
#if __has_builtin(__builtin_amdgcn_tensor_load_to_lds) && \
    __has_builtin(__builtin_amdgcn_s_wait_tensorcnt)
  if (threadIdx.x < 32) {  // single wave issues the DMA (EXEC ignored by TDM)
    uint64_t ga = (uint64_t)(uintptr_t)gsrc;
    uint32_t la = lds_offset(ldst);
    u32x4 g0;
    g0[0] = 1u;                                              // count=1 (valid D#)
    g0[1] = la;                                              // lds_addr
    g0[2] = (uint32_t)ga;                                    // global_addr[31:0]
    g0[3] = (uint32_t)((ga >> 32) & 0x01FFFFFFu) | (2u << 30); // addr[56:32]|type=2
    i32x8 g1;
    g1[0] = (int)(1u << 16);                                 // data_size=1 -> 2B
    g1[1] = (int)(((uint32_t)rowlen & 0xFFFFu) << 16);       // tensor_dim0[15:0]
    g1[2] = (int)((((uint32_t)rowlen >> 16) & 0xFFFFu)       // tensor_dim0[31:16]
                  | (((uint32_t)nrows & 0xFFFFu) << 16));    // tensor_dim1[15:0]
    g1[3] = (int)(((uint32_t)rowlen & 0xFFFFu) << 16);       // td1 hi=0 | tile_dim0
    g1[4] = (int)((uint32_t)nrows & 0xFFFFu);                // tile_dim1 | tile_dim2=0
    g1[5] = rowlen;                                          // tensor_dim0_stride lo
    g1[6] = 0;                                               // stride0 hi | stride1 lo
    g1[7] = 0;                                               // stride1 hi
    i32x4 gz4 = {0, 0, 0, 0};
    i32x8 gz8 = {0, 0, 0, 0, 0, 0, 0, 0};
    __builtin_amdgcn_tensor_load_to_lds(g0, g1, gz4, gz4, gz8, 0);
    __builtin_amdgcn_s_wait_tensorcnt(0);
  }
#else
  const uint4* src = (const uint4*)gsrc;
  uint4* dst = (uint4*)ldst;
  int chunks = (rowlen * nrows) >> 3;  // bf16 -> 16B chunks
  for (int i = threadIdx.x; i < chunks; i += blockDim.x) dst[i] = src[i];
#endif
}

// ---------------- device-wide barrier (all blocks resident) ----------------
__device__ __forceinline__ void grid_barrier(int* cnt, int* gen, int nblocks) {
  __syncthreads();
  if (threadIdx.x == 0) {
    __threadfence();
    int g = __hip_atomic_load(gen, __ATOMIC_RELAXED, __HIP_MEMORY_SCOPE_AGENT);
    int arrived = __hip_atomic_fetch_add(cnt, 1, __ATOMIC_ACQ_REL,
                                         __HIP_MEMORY_SCOPE_AGENT);
    if (arrived == nblocks - 1) {
      __hip_atomic_store(cnt, 0, __ATOMIC_RELAXED, __HIP_MEMORY_SCOPE_AGENT);
      __hip_atomic_fetch_add(gen, 1, __ATOMIC_ACQ_REL, __HIP_MEMORY_SCOPE_AGENT);
    } else {
      while (__hip_atomic_load(gen, __ATOMIC_ACQUIRE, __HIP_MEMORY_SCOPE_AGENT) == g) {
        __builtin_amdgcn_s_sleep(8);
      }
    }
  }
  __syncthreads();
}

// ---------------- kernel 1: pack fused bf16 weights, zero barrier ----------------
__global__ void prep_kernel(char* ws, const float* Wih1, const float* Whh1,
                            const float* bih1, const float* bhh1,
                            const float* Wih2, const float* Whh2,
                            const float* bih2, const float* bhh2) {
  __bf16* wc1 = (__bf16*)(ws + OFF_WCAT1);
  __bf16* wc2 = (__bf16*)(ws + OFF_WCAT2);
  float* b1 = (float*)(ws + OFF_B1);
  float* b2 = (float*)(ws + OFF_B2);
  int* bar = (int*)(ws + OFF_BAR);
  int tid = blockIdx.x * blockDim.x + threadIdx.x;
  int nt = gridDim.x * blockDim.x;
  for (int i = tid; i < N1 * K1; i += nt) {
    int n = i / K1, k = i - n * K1;
    float v = (k < Ez + Pz) ? Wih1[(size_t)n * (Ez + Pz) + k]
                            : Whh1[(size_t)n * Hz + (k - (Ez + Pz))];
    wc1[i] = (__bf16)v;
  }
  for (int i = tid; i < N2 * K2; i += nt) {
    int n = i / K2, k = i - n * K2;
    float v = (k < Hz) ? Wih2[(size_t)n * Hz + k]
                       : Whh2[(size_t)n * Dz + (k - Hz)];
    wc2[i] = (__bf16)v;
  }
  for (int i = tid; i < N1; i += nt) b1[i] = bih1[i] + bhh1[i];
  for (int i = tid; i < N2; i += nt) b2[i] = bih2[i] + bhh2[i];
  if (tid == 0) { bar[0] = 0; bar[1] = 0; }
}

// ---------------- kernel 2: K/V projection (shared-A dual WMMA GEMM) ----------------
__global__ void kv_kernel(char* ws, const float* enc, const float* Wk,
                          const float* bk, const float* Wv, const float* bv) {
  float* keys = (float*)(ws + OFF_KEYS);
  float* vals = (float*)(ws + OFF_VALS);
  int tid = blockIdx.x * blockDim.x + threadIdx.x;
  int gw = tid >> 5, lane = tid & 31;
  int nw = (gridDim.x * blockDim.x) >> 5;
  const int MT = (Bz * Sz) / 16;  // 3200
  const int NT = Pz / 16;         // 8
  for (int tile = gw; tile < MT * NT; tile += nw) {
    int tn = tile & (NT - 1), tm = tile >> 3;
    int m0 = tm * 16, n0 = tn * 16;
    v8f ak = {}, av = {};
    for (int k0 = 0; k0 < ENCz; k0 += 32) {
      if (k0 + 32 < ENCz)
        __builtin_prefetch(enc + (size_t)m0 * ENCz + k0 + 32, 0, 1);
      v16bf a = load_a_f32(enc, ENCz, m0, k0, lane);
      v16bf bkf = load_b_wf32(Wk, ENCz, n0, k0, lane);
      v16bf bvf = load_b_wf32(Wv, ENCz, n0, k0, lane);
      ak = wmma_bf16(a, bkf, ak);
      av = wmma_bf16(a, bvf, av);
    }
    int lo = lane & 15, hi = lane >> 4;
#pragma unroll
    for (int r = 0; r < 8; ++r) {
      int row = m0 + r + 8 * hi, col = n0 + lo;
      keys[(size_t)row * Pz + col] = ak[r] + bk[col];
      vals[(size_t)row * Pz + col] = av[r] + bv[col];
    }
  }
}

// ---------------- kernel 3: persistent sequential decoder ----------------
// 128 blocks x 256 threads. Block b owns batch row b; its recurrent state
// lives in LDS for the whole kernel. Only WMMA GEMM operands (xcat, g1,
// xcat2, g2) cross blocks via global memory. 4 grid barriers per timestep.
__global__ void __launch_bounds__(256)
decode_kernel(char* ws, const float* embedw, const float* Wq, const float* bq,
              const float* b_char, const int* lens, float* out) {
  const float* keys = (const float*)(ws + OFF_KEYS);
  const float* vals = (const float*)(ws + OFF_VALS);
  const __bf16* wc1 = (const __bf16*)(ws + OFF_WCAT1);
  const __bf16* wc2 = (const __bf16*)(ws + OFF_WCAT2);
  const float* b1 = (const float*)(ws + OFF_B1);
  const float* b2 = (const float*)(ws + OFF_B2);
  __bf16* xcat = (__bf16*)(ws + OFF_XCAT);
  __bf16* xcat2 = (__bf16*)(ws + OFF_XCAT2);
  float* g1 = (float*)(ws + OFF_G1);
  float* g2 = (float*)(ws + OFF_G2);
  int* bar = (int*)(ws + OFF_BAR);

  const int b = blockIdx.x;              // batch row owned by this block
  const int NB = gridDim.x;              // 128
  const int wv = threadIdx.x >> 5;       // wave in block: 0..7 (= m-tile)
  const int lane = threadIdx.x & 31;
  const float scale = 0.08838834764831845f;  // 1/sqrt(128)
  const size_t ATTN = (size_t)Bz * Tz * Vz;

  __shared__ __align__(16) __bf16 s_w[16 * K1];   // 28672 B weight slab (TDM dest)
  __shared__ float s_h1[Hz], s_c1[Hz];
  __shared__ float s_h2[Dz], s_c2[Dz], s_ctx[Dz], s_q[Pz];
  __shared__ float s_sc[Sz];
  __shared__ float s_red[256];
  __shared__ float s_pv[32];
  __shared__ int s_char;

  // ---- init recurrent state ----
  for (int i = threadIdx.x; i < Hz; i += blockDim.x) { s_h1[i] = 0.f; s_c1[i] = 0.f; }
  for (int i = threadIdx.x; i < Dz; i += blockDim.x) {
    s_h2[i] = 0.f; s_c2[i] = 0.f; s_ctx[i] = 0.f;
  }
  if (threadIdx.x == 0) s_char = 0;  // SOS
  __syncthreads();

  // ---- P1(t=0): xcat row b = bf16([embed[char], ctx, h1]) ----
  for (int k = threadIdx.x; k < K1; k += blockDim.x) {
    float v = (k < Ez) ? embedw[(size_t)s_char * Ez + k]
            : (k < Ez + Pz) ? s_ctx[k - Ez] : s_h1[k - Ez - Pz];
    xcat[(size_t)b * K1 + k] = (__bf16)v;
  }

  const int len = lens[b];

  for (int t = 0; t < Tz; ++t) {
    grid_barrier(bar, bar + 1, NB);  // xcat complete

    // ---- P2: gates1 = xcat @ Wcat1^T (WMMA; block = n-tile, wave = m-tile) ----
    {
      const int n0 = b * 16;  // this block's 16 output columns
      stage_rows_lds(wc1 + (size_t)n0 * K1, s_w, K1, 16);  // TDM -> LDS
      __syncthreads();
      int m0 = wv * 16;
      v8f acc = {};
      for (int k0 = 0; k0 < K1; k0 += 32) {
        v16bf a = load_a_bf16(xcat, K1, m0, k0, lane);
        v16bf bb = load_b_rows(s_w, K1, k0, lane);   // ds_load path
        acc = wmma_bf16(a, bb, acc);
      }
      int lo = lane & 15, hi = lane >> 4;
#pragma unroll
      for (int r = 0; r < 8; ++r)
        g1[(size_t)(m0 + r + 8 * hi) * N1 + n0 + lo] = acc[r];
    }
    grid_barrier(bar, bar + 1, NB);  // g1 complete

    // ---- P3: LSTM1 pointwise for row b (bias folded); build xcat2 row ----
    {
      const float* g = g1 + (size_t)b * N1;
      for (int d = threadIdx.x; d < Hz; d += blockDim.x) {
        float gi = g[d] + b1[d];
        float gf = g[Hz + d] + b1[Hz + d];
        float gg = g[2 * Hz + d] + b1[2 * Hz + d];
        float go = g[3 * Hz + d] + b1[3 * Hz + d];
        float c = sigf(gf) * s_c1[d] + sigf(gi) * tanhf(gg);
        float h = sigf(go) * tanhf(c);
        s_c1[d] = c; s_h1[d] = h;
        xcat2[(size_t)b * K2 + d] = (__bf16)h;
      }
      for (int d = threadIdx.x; d < Dz; d += blockDim.x)
        xcat2[(size_t)b * K2 + Hz + d] = (__bf16)s_h2[d];  // h2_{t-1}
    }
    grid_barrier(bar, bar + 1, NB);  // xcat2 complete

    // ---- P4: gates2 = xcat2 @ Wcat2^T (WMMA; blocks 0..31 = n-tiles) ----
    if (b < N2 / 16) {
      const int n0 = b * 16;
      stage_rows_lds(wc2 + (size_t)n0 * K2, s_w, K2, 16);  // TDM -> LDS
      __syncthreads();
      int m0 = wv * 16;
      v8f acc = {};
      for (int k0 = 0; k0 < K2; k0 += 32) {
        v16bf a = load_a_bf16(xcat2, K2, m0, k0, lane);
        v16bf bb = load_b_rows(s_w, K2, k0, lane);
        acc = wmma_bf16(a, bb, acc);
      }
      int lo = lane & 15, hi = lane >> 4;
#pragma unroll
      for (int r = 0; r < 8; ++r)
        g2[(size_t)(m0 + r + 8 * hi) * N2 + n0 + lo] = acc[r];
    }
    grid_barrier(bar, bar + 1, NB);  // g2 complete

    // ---- P5 (fused, per-block): LSTM2, query, attention, logits, argmax ----
    {
      if (threadIdx.x < Dz) {
        int d = threadIdx.x;
        const float* g = g2 + (size_t)b * N2;
        float gi = g[d] + b2[d];
        float gf = g[Dz + d] + b2[Dz + d];
        float gg = g[2 * Dz + d] + b2[2 * Dz + d];
        float go = g[3 * Dz + d] + b2[3 * Dz + d];
        float c = sigf(gf) * s_c2[d] + sigf(gi) * tanhf(gg);
        float h = sigf(go) * tanhf(c);
        s_c2[d] = c; s_h2[d] = h;
      }
      __syncthreads();
      if (threadIdx.x < Pz) {   // q = h2 @ Wq^T + bq (tiny, VALU)
        int p = threadIdx.x;
        const float* wr = Wq + (size_t)p * Dz;
        float acc = bq[p];
#pragma unroll 4
        for (int d = 0; d < Dz; ++d) acc += s_h2[d] * wr[d];
        s_q[p] = acc;
      }
      __syncthreads();
      for (int s = threadIdx.x; s < Sz; s += blockDim.x) {  // masked scores
        const float* kp = keys + ((size_t)b * Sz + s) * Pz;
        float acc = 0.f;
#pragma unroll 4
        for (int p = 0; p < Pz; ++p) acc += kp[p] * s_q[p];
        s_sc[s] = (s < len) ? acc * scale : -1e9f;
      }
      __syncthreads();
      float mx = -1e30f;  // softmax max
      for (int s = threadIdx.x; s < Sz; s += blockDim.x) mx = fmaxf(mx, s_sc[s]);
      s_red[threadIdx.x] = mx; __syncthreads();
      for (int off = 128; off > 0; off >>= 1) {
        if (threadIdx.x < off)
          s_red[threadIdx.x] = fmaxf(s_red[threadIdx.x], s_red[threadIdx.x + off]);
        __syncthreads();
      }
      mx = s_red[0]; __syncthreads();
      float sm = 0.f;  // exp + sum
      for (int s = threadIdx.x; s < Sz; s += blockDim.x) {
        float w = __expf(s_sc[s] - mx);
        s_sc[s] = w; sm += w;
      }
      s_red[threadIdx.x] = sm; __syncthreads();
      for (int off = 128; off > 0; off >>= 1) {
        if (threadIdx.x < off) s_red[threadIdx.x] += s_red[threadIdx.x + off];
        __syncthreads();
      }
      float inv = 1.0f / s_red[0]; __syncthreads();
      if (threadIdx.x < Pz) {  // context
        int p = threadIdx.x;
        float acc = 0.f;
        const float* vp = vals + ((size_t)b * Sz) * Pz + p;
        for (int s = 0; s < Sz; ++s) acc += s_sc[s] * vp[(size_t)s * Pz];
        s_ctx[p] = acc * inv;
      }
      if (b == 0)  // attention plot for batch 0
        for (int s = threadIdx.x; s < Sz; s += blockDim.x)
          out[ATTN + (size_t)t * Sz + s] = s_sc[s] * inv;
      __syncthreads();
      if (threadIdx.x < Vz) {  // logits over tied embedding
        int v = threadIdx.x;
        float acc = b_char[v];
        const float* er = embedw + (size_t)v * Ez;
#pragma unroll 4
        for (int k = 0; k < Pz; ++k) acc += s_q[k] * er[k];
#pragma unroll 4
        for (int k = 0; k < Pz; ++k) acc += s_ctx[k] * er[Pz + k];
        s_pv[v] = acc;
        out[(size_t)b * Tz * Vz + (size_t)t * Vz + v] = acc;
      }
      __syncthreads();
      if (threadIdx.x == 0) {  // argmax (first-max tie rule)
        float best = s_pv[0]; int bi = 0;
        for (int w = 1; w < Vz; ++w)
          if (s_pv[w] > best) { best = s_pv[w]; bi = w; }
        s_char = bi;
      }
      __syncthreads();
    }

    // ---- P1(t+1): next xcat row (block-local deps only; no grid barrier) ----
    for (int k = threadIdx.x; k < K1; k += blockDim.x) {
      float v = (k < Ez) ? embedw[(size_t)s_char * Ez + k]
              : (k < Ez + Pz) ? s_ctx[k - Ez] : s_h1[k - Ez - Pz];
      xcat[(size_t)b * K1 + k] = (__bf16)v;
    }
  }
}

// ---------------- host entry ----------------
extern "C" void kernel_launch(void* const* d_in, const int* in_sizes, int n_in,
                              void* d_out, int out_size, void* d_ws, size_t ws_size,
                              hipStream_t stream) {
  (void)in_sizes; (void)n_in; (void)out_size; (void)ws_size;
  const float* enc   = (const float*)d_in[0];
  const int*   lens  = (const int*)  d_in[1];
  const float* embed = (const float*)d_in[2];
  const float* Wih1  = (const float*)d_in[3];
  const float* Whh1  = (const float*)d_in[4];
  const float* bih1  = (const float*)d_in[5];
  const float* bhh1  = (const float*)d_in[6];
  const float* Wih2  = (const float*)d_in[7];
  const float* Whh2  = (const float*)d_in[8];
  const float* bih2  = (const float*)d_in[9];
  const float* bhh2  = (const float*)d_in[10];
  const float* Wk    = (const float*)d_in[11];
  const float* bk    = (const float*)d_in[12];
  const float* Wv    = (const float*)d_in[13];
  const float* bv    = (const float*)d_in[14];
  const float* Wq    = (const float*)d_in[15];
  const float* bq    = (const float*)d_in[16];
  const float* bch   = (const float*)d_in[17];
  float* out = (float*)d_out;
  char* ws = (char*)d_ws;

  prep_kernel<<<256, 256, 0, stream>>>(ws, Wih1, Whh1, bih1, bhh1, Wih2, Whh2,
                                       bih2, bhh2);
  kv_kernel<<<3200, 256, 0, stream>>>(ws, enc, Wk, bk, Wv, bv);
  decode_kernel<<<128, 256, 0, stream>>>(ws, embed, Wq, bq, bch, lens, out);
}
